// TransformerSelfAttnLayer_61529701483239
// MI455X (gfx1250) — matrix operand order repaired and checked by reading, not compiled, not run on hardware
//
#include <hip/hip_runtime.h>
#include <hip/hip_bf16.h>
#include <math.h>

// ---------------- problem constants ----------------
#define BS    4
#define SEQ   32768           // 128*256
#define CDIM  128
#define NH    8
#define NTOK  (BS*SEQ)        // 131072 tokens
#define NBLK  (NTOK/128)      // 1024 blocks of 128 tokens
#define BLK_PER_B (SEQ/128)   // 256 blocks per batch
#define LDS_STRIDE 136        // halves per row (272B: 16B aligned, conflict-free)

// ---------------- WMMA types / helpers ----------------
typedef _Float16 v16h __attribute__((ext_vector_type(16)));
typedef _Float16 h8   __attribute__((ext_vector_type(8)));
typedef _Float16 h4   __attribute__((ext_vector_type(4)));
typedef float    v8f  __attribute__((ext_vector_type(8)));
typedef unsigned int u32x4 __attribute__((ext_vector_type(4)));
typedef int      i32x4 __attribute__((ext_vector_type(4)));
typedef int      i32x8 __attribute__((ext_vector_type(8)));

__device__ __forceinline__ v16h load_bfrag(const _Float16* p) {
  // B fragment: K = g*8+e (e<8), g*8+16+(e-8), matching the A fragment (p = base + g*16)
  int boff = ((threadIdx.x >> 4) & 1) * 8;
  h8 lo = *(const h8*)(p - boff);
  h8 hi = *(const h8*)(p + 16 - boff);
  return __builtin_shufflevector(lo, hi, 0,1,2,3,4,5,6,7,8,9,10,11,12,13,14,15);
}
__device__ __forceinline__ v16h load_afrag(const _Float16* p) {
  // A fragment: per ISA 16-bit A 16x32 layout: K = g*8+e (e<8), g*8+16+(e-8)
  h8 lo = *(const h8*)p;
  h8 hi = *(const h8*)(p + 16);
  return __builtin_shufflevector(lo, hi, 0,1,2,3,4,5,6,7,8,9,10,11,12,13,14,15);
}
__device__ __forceinline__ v8f wmma16(v16h a, v16h b, v8f c) {
  return __builtin_amdgcn_wmma_f32_16x16x32_f16(false, a, false, b, (short)0, c, false, false);
}

// ---- Tensor Data Mover: stage `rows` x 128-half matrix into LDS with padded
// ---- rows (stride 136 halves). D# built per CDNA5 ISA 8.3/8.4.
__device__ __forceinline__ void tdm_load_w(unsigned int lds_addr, const _Float16* gsrc,
                                           unsigned int rows) {
  unsigned long long ga = (unsigned long long)(uintptr_t)gsrc;
  u32x4 g0;
  g0[0] = 1u;                                         // count=1 valid, user mode
  g0[1] = lds_addr;                                   // lds_addr (bytes)
  g0[2] = (unsigned int)ga;                           // global_addr[31:0]
  g0[3] = (unsigned int)((ga >> 32) & 0x01ffffffu) | (2u << 30);  // addr[56:32], type=2
  i32x8 g1;
  // data_size=1(2B), pad_enable, pad_interval=5(64 dw = one row), pad_amount=3(4 dw = 16B)
  g1[0] = (int)((1u << 16) | (1u << 20) | (5u << 22) | (3u << 25));
  g1[1] = (int)(128u << 16);                          // tensor_dim0=128 -> bits[79:48] lo16
  g1[2] = (int)(rows << 16);                          // dim0 hi=0 | tensor_dim1 lo16
  g1[3] = (int)(128u << 16);                          // dim1 hi=0 | tile_dim0=128
  g1[4] = (int)(rows & 0xffffu);                      // tile_dim1=rows, tile_dim2=0
  g1[5] = (int)128u;                                  // tensor_dim0_stride = 128 (lo32)
  g1[6] = 0;                                          // stride0 hi | stride1 lo
  g1[7] = 0;
  i32x4 z4 = {};
#if __has_include(<hip/amd_detail/amd_gfx1250_TDM.h>)
  i32x8 z8 = {};
  __builtin_amdgcn_tensor_load_to_lds(g0, g1, z4, z4, z8, 0);
#else
  __builtin_amdgcn_tensor_load_to_lds(g0, g1, z4, z4, 0);
#endif
}

// ---------------- workspace layout (bytes) ----------------
static constexpr size_t OFF_WQKV = 0;                                     // 3*16384 f16
static constexpr size_t OFF_W1   = OFF_WQKV + (size_t)3*16384*2;
static constexpr size_t OFF_W2   = OFF_W1   + (size_t)16384*2;
static constexpr size_t OFF_XN   = 262144;                                // xn f16 (later: mn)
static constexpr size_t OFF_Q    = OFF_XN   + (size_t)NTOK*CDIM*2;        // Q f16 (later: ffn hidden)
static constexpr size_t OFF_PART = OFF_Q    + (size_t)NTOK*CDIM*2;        // per-block KV/Ksum partials
static constexpr size_t OFF_KV   = OFF_PART + (size_t)NBLK*NH*272*4;
static constexpr size_t OFF_KSUM = OFF_KV   + (size_t)BS*NH*256*4;
static constexpr size_t OFF_CMAT = OFF_KSUM + (size_t)BS*NH*16*4;
static constexpr size_t OFF_MSG  = OFF_CMAT + (size_t)BS*16384*2;         // msg f32

// ---------------- kernels ----------------

// weights f32 -> f16 (Wq,Wk,Wv,W1,W2 concatenated, row-major [out,in])
__global__ void cvt_w_kernel(const float* __restrict__ wq, const float* __restrict__ wk,
                             const float* __restrict__ wv, const float* __restrict__ w1,
                             const float* __restrict__ w2, _Float16* __restrict__ dst) {
  int idx = blockIdx.x * blockDim.x + threadIdx.x;   // 0..81919
  int seg = idx >> 14, off = idx & 16383;
  const float* src = (seg == 0) ? wq : (seg == 1) ? wk : (seg == 2) ? wv : (seg == 3) ? w1 : w2;
  dst[idx] = (_Float16)src[off];
}

// LayerNorm over c=128, wave-per-token, f32 in -> f16 out
__global__ __launch_bounds__(256) void ln_kernel(const float* __restrict__ in,
                                                 const float* __restrict__ gam,
                                                 const float* __restrict__ bet,
                                                 _Float16* __restrict__ out) {
  int wave = threadIdx.x >> 5, lane = threadIdx.x & 31;
  size_t tok = (size_t)blockIdx.x * 8 + wave;
  const float4 xv = ((const float4*)(in + tok * CDIM))[lane];
  float s = xv.x + xv.y + xv.z + xv.w;
  #pragma unroll
  for (int m = 16; m >= 1; m >>= 1) s += __shfl_xor(s, m, 32);
  float mu = s * (1.f / 128.f);
  float d0 = xv.x - mu, d1 = xv.y - mu, d2 = xv.z - mu, d3 = xv.w - mu;
  float v = d0*d0 + d1*d1 + d2*d2 + d3*d3;
  #pragma unroll
  for (int m = 16; m >= 1; m >>= 1) v += __shfl_xor(v, m, 32);
  float rs = rsqrtf(v * (1.f / 128.f) + 1e-5f);
  const float4 gv = ((const float4*)gam)[lane];
  const float4 bv = ((const float4*)bet)[lane];
  h4 o;
  o[0] = (_Float16)(d0 * rs * gv.x + bv.x);
  o[1] = (_Float16)(d1 * rs * gv.y + bv.y);
  o[2] = (_Float16)(d2 * rs * gv.z + bv.z);
  o[3] = (_Float16)(d3 * rs * gv.w + bv.w);
  ((h4*)(out + tok * CDIM))[lane] = o;
}

// QKV GEMM (WMMA, weights TDM-staged in LDS) + phi + per-block KV/Ksum partials
__global__ __launch_bounds__(256) void qkv_kernel(const _Float16* __restrict__ xn,
                                                  const _Float16* __restrict__ wqkv,
                                                  _Float16* __restrict__ qout,
                                                  float* __restrict__ part) {
  extern __shared__ _Float16 smem[];
  _Float16* sW  = smem;                          // 384 rows x stride 136 (TDM-padded)
  _Float16* sKT = smem + 384 * LDS_STRIDE;       // [128 c][128 tok]
  _Float16* sVT = sKT  + 128 * LDS_STRIDE;
  int wave = threadIdx.x >> 5, lane = threadIdx.x & 31;
  int g = lane >> 4, ln = lane & 15;
  size_t tok0 = (size_t)blockIdx.x * 128 + wave * 16;

  if (threadIdx.x < 32)                          // wave 0 drives the TDM
    tdm_load_w((unsigned int)(uintptr_t)sW, wqkv, 384);

  // overlap: every wave loads its A fragments from global while TDM runs
  v16h afr[4];
  const _Float16* arow = xn + (tok0 + ln) * CDIM;
  #pragma unroll
  for (int ks = 0; ks < 4; ++ks) afr[ks] = load_afrag(arow + ks * 32 + g * 8);

  if (threadIdx.x < 32) __builtin_amdgcn_s_wait_tensorcnt(0);
  __syncthreads();

  // ---- Q = elu(xn@Wq^T)+1 -> global ----
  for (int nt = 0; nt < 8; ++nt) {
    const _Float16* brow = sW + (size_t)(0 * 128 + nt * 16 + ln) * LDS_STRIDE;
    v16h bfr[4];
    #pragma unroll
    for (int ks = 0; ks < 4; ++ks) bfr[ks] = load_bfrag(brow + ks * 32 + g * 16);
    v8f acc = {};
    #pragma unroll
    for (int ks = 0; ks < 4; ++ks) acc = wmma16(afr[ks], bfr[ks], acc);
    #pragma unroll
    for (int r = 0; r < 8; ++r) {
      float x = acc[r];
      float qv = x > 0.f ? x + 1.f : __expf(x);
      qout[(tok0 + r + g * 8) * CDIM + nt * 16 + ln] = (_Float16)qv;
    }
  }
  // ---- K = elu(xn@Wk^T)+1 -> LDS transposed ----
  for (int nt = 0; nt < 8; ++nt) {
    const _Float16* brow = sW + (size_t)(1 * 128 + nt * 16 + ln) * LDS_STRIDE;
    v16h bfr[4];
    #pragma unroll
    for (int ks = 0; ks < 4; ++ks) bfr[ks] = load_bfrag(brow + ks * 32 + g * 16);
    v8f acc = {};
    #pragma unroll
    for (int ks = 0; ks < 4; ++ks) acc = wmma16(afr[ks], bfr[ks], acc);
    #pragma unroll
    for (int r = 0; r < 8; ++r) {
      float x = acc[r];
      float kv = x > 0.f ? x + 1.f : __expf(x);
      sKT[(nt * 16 + ln) * LDS_STRIDE + (wave * 16 + r + g * 8)] = (_Float16)kv;
    }
  }
  // ---- V/s -> LDS transposed ----
  for (int nt = 0; nt < 8; ++nt) {
    const _Float16* brow = sW + (size_t)(2 * 128 + nt * 16 + ln) * LDS_STRIDE;
    v16h bfr[4];
    #pragma unroll
    for (int ks = 0; ks < 4; ++ks) bfr[ks] = load_bfrag(brow + ks * 32 + g * 16);
    v8f acc = {};
    #pragma unroll
    for (int ks = 0; ks < 4; ++ks) acc = wmma16(afr[ks], bfr[ks], acc);
    #pragma unroll
    for (int r = 0; r < 8; ++r)
      sVT[(nt * 16 + ln) * LDS_STRIDE + (wave * 16 + r + g * 8)] =
          (_Float16)(acc[r] * (1.f / 32768.f));
  }
  __syncthreads();

  // phase 2: wave == head. KV[h] = K^T(16x128tok) @ V(128tok x 16) via 4 WMMA
  {
    int hh = wave;
    v8f acc = {};
    const _Float16* ka = sKT + (size_t)(hh * 16 + ln) * LDS_STRIDE;  // A rows = d
    const _Float16* vb = sVT + (size_t)(hh * 16 + ln) * LDS_STRIDE;  // B cols = v'
    #pragma unroll
    for (int ks = 0; ks < 4; ++ks)
      acc = wmma16(load_afrag(ka + ks * 32 + g * 8),
                   load_bfrag(vb + ks * 32 + g * 16), acc);
    float* pkv = part + ((size_t)blockIdx.x * NH + hh) * 272;
    #pragma unroll
    for (int r = 0; r < 8; ++r)
      pkv[(r + g * 8) * 16 + ln] = acc[r];
    // Ksum[h,d] over this block's 128 tokens
    float s = 0.f;
    const _Float16* kr = sKT + (size_t)(hh * 16 + ln) * LDS_STRIDE + g * 64;
    for (int t = 0; t < 64; ++t) s += (float)kr[t];
    s += __shfl_xor(s, 16, 32);
    if (g == 0) pkv[256 + ln] = s;
  }
}

// deterministic fixed-order reduction of block partials
__global__ void reduce_kv_kernel(const float* __restrict__ part,
                                 float* __restrict__ kv, float* __restrict__ ksum) {
  int b = blockIdx.x >> 3, h = blockIdx.x & 7;
  int tid = threadIdx.x;
  const float* base = part + ((size_t)b * BLK_PER_B * NH + h) * 272;
  if (tid < 256) {
    float s = 0.f;
    for (int p = 0; p < BLK_PER_B; ++p) s += base[(size_t)p * NH * 272 + tid];
    kv[(b * NH + h) * 256 + tid] = s;
  } else if (tid < 272) {
    int d = tid - 256;
    float s = 0.f;
    for (int p = 0; p < BLK_PER_B; ++p) s += base[(size_t)p * NH * 272 + 256 + d];
    ksum[(b * NH + h) * 16 + d] = s;
  }
}

// Cmat_b[j][c] = sum_v KV[b, c/16, c%16, v] * Wm[j, (c/16)*16+v]  (f16, [j][c] row-major)
__global__ void make_cmat_kernel(const float* __restrict__ kv, const float* __restrict__ wm,
                                 _Float16* __restrict__ cmat) {
  int b = blockIdx.x;
  for (int idx = threadIdx.x; idx < 16384; idx += blockDim.x) {
    int j = idx >> 7, c = idx & 127;
    int hh = c >> 4, d = c & 15;
    const float* kvp = kv + ((size_t)(b * NH + hh) * 256 + d * 16);
    const float* wmp = wm + (size_t)j * CDIM + hh * 16;
    float s = 0.f;
    #pragma unroll
    for (int v = 0; v < 16; ++v) s += kvp[v] * wmp[v];
    cmat[(size_t)b * 16384 + idx] = (_Float16)s;
  }
}

// Z-scale Q, GEMM vs TDM-staged Cmat, add residual -> msg (f32), then LN2 -> mn (f16)
__global__ __launch_bounds__(256) void msg_kernel(const _Float16* __restrict__ q,
                                                  const float* __restrict__ ksum,
                                                  const _Float16* __restrict__ cmat,
                                                  const float* __restrict__ x,
                                                  const float* __restrict__ g2,
                                                  const float* __restrict__ b2,
                                                  float* __restrict__ msg,
                                                  _Float16* __restrict__ mn) {
  extern __shared__ _Float16 smem2[];
  _Float16* sQz = smem2;                         // [128 tok][stride 136]
  _Float16* sCb = smem2 + 128 * LDS_STRIDE;      // staged Cmat for this batch
  int batch = blockIdx.x / BLK_PER_B;
  size_t tok0 = (size_t)blockIdx.x * 128;
  int tid = threadIdx.x;

  if (tid < 32)
    tdm_load_w((unsigned int)(uintptr_t)sCb, cmat + (size_t)batch * 16384, 128);

  // phase A (overlaps TDM): per (token, head) Z = s/(Q.Ksum + eps); Qz -> LDS
  {
    int tt = tid >> 1, half = tid & 1;
    const _Float16* qr = q + (tok0 + tt) * CDIM + half * 64;
    const float* ks = ksum + (size_t)(batch * NH + half * 4) * 16;
    _Float16* dst = sQz + tt * LDS_STRIDE + half * 64;
    #pragma unroll
    for (int hh = 0; hh < 4; ++hh) {
      h8 lo = *(const h8*)(qr + hh * 16);
      h8 hi = *(const h8*)(qr + hh * 16 + 8);
      float a[16];
      #pragma unroll
      for (int d = 0; d < 8; ++d) { a[d] = (float)lo[d]; a[d + 8] = (float)hi[d]; }
      float dot = 0.f;
      #pragma unroll
      for (int d = 0; d < 16; ++d) dot += a[d] * ks[hh * 16 + d];
      float z = 32768.f / (dot + 1e-6f);
      #pragma unroll
      for (int d = 0; d < 16; ++d) dst[hh * 16 + d] = (_Float16)(a[d] * z);
    }
  }
  if (tid < 32) __builtin_amdgcn_s_wait_tensorcnt(0);
  __syncthreads();

  // phase B: msg = x + Qz @ Cmat_b^T  (WMMA, A and B both from LDS)
  int wave = tid >> 5, lane = tid & 31, g = lane >> 4, ln = lane & 15;
  {
    const _Float16* arow = sQz + (size_t)(wave * 16 + ln) * LDS_STRIDE;
    v16h afr[4];
    #pragma unroll
    for (int ks2 = 0; ks2 < 4; ++ks2) afr[ks2] = load_afrag(arow + ks2 * 32 + g * 8);
    for (int nt = 0; nt < 8; ++nt) {
      const _Float16* brow = sCb + (size_t)(nt * 16 + ln) * LDS_STRIDE;
      v16h bfr[4];
      #pragma unroll
      for (int ks2 = 0; ks2 < 4; ++ks2) bfr[ks2] = load_bfrag(brow + ks2 * 32 + g * 16);
      v8f acc = {};
      #pragma unroll
      for (int ks2 = 0; ks2 < 4; ++ks2) acc = wmma16(afr[ks2], bfr[ks2], acc);
      #pragma unroll
      for (int r = 0; r < 8; ++r) {
        size_t t = tok0 + wave * 16 + r + g * 8;
        int col = nt * 16 + ln;
        msg[t * CDIM + col] = acc[r] + x[t * CDIM + col];
      }
    }
  }
  __syncthreads();   // make msg stores visible block-wide before LN reads

  // phase C: LN2 over this wave's 16 tokens -> mn f16
  {
    const float4 gv = ((const float4*)g2)[lane];
    const float4 bv = ((const float4*)b2)[lane];
    for (int i = 0; i < 16; ++i) {
      size_t t = tok0 + wave * 16 + i;
      const float4 xv = ((const float4*)(msg + t * CDIM))[lane];
      float s = xv.x + xv.y + xv.z + xv.w;
      #pragma unroll
      for (int m = 16; m >= 1; m >>= 1) s += __shfl_xor(s, m, 32);
      float mu = s * (1.f / 128.f);
      float d0 = xv.x - mu, d1 = xv.y - mu, d2 = xv.z - mu, d3 = xv.w - mu;
      float v = d0*d0 + d1*d1 + d2*d2 + d3*d3;
      #pragma unroll
      for (int m = 16; m >= 1; m >>= 1) v += __shfl_xor(v, m, 32);
      float rs = rsqrtf(v * (1.f / 128.f) + 1e-5f);
      h4 o;
      o[0] = (_Float16)(d0 * rs * gv.x + bv.x);
      o[1] = (_Float16)(d1 * rs * gv.y + bv.y);
      o[2] = (_Float16)(d2 * rs * gv.z + bv.z);
      o[3] = (_Float16)(d3 * rs * gv.w + bv.w);
      ((h4*)(mn + t * CDIM))[lane] = o;
    }
  }
}

// FFN GEMM1 (W1 TDM-staged) + exact GELU -> f16
__global__ __launch_bounds__(256) void ffn1_kernel(const _Float16* __restrict__ mn,
                                                   const _Float16* __restrict__ w1,
                                                   _Float16* __restrict__ hout) {
  extern __shared__ _Float16 sW1[];              // 128 x stride 136
  int wave = threadIdx.x >> 5, lane = threadIdx.x & 31, g = lane >> 4, ln = lane & 15;
  size_t tok0 = (size_t)blockIdx.x * 128 + wave * 16;
  if (threadIdx.x < 32) tdm_load_w((unsigned int)(uintptr_t)sW1, w1, 128);
  const _Float16* arow = mn + (tok0 + ln) * CDIM;
  v16h afr[4];
  #pragma unroll
  for (int ks = 0; ks < 4; ++ks) afr[ks] = load_afrag(arow + ks * 32 + g * 8);
  if (threadIdx.x < 32) __builtin_amdgcn_s_wait_tensorcnt(0);
  __syncthreads();
  for (int nt = 0; nt < 8; ++nt) {
    const _Float16* brow = sW1 + (size_t)(nt * 16 + ln) * LDS_STRIDE;
    v16h bfr[4];
    #pragma unroll
    for (int ks = 0; ks < 4; ++ks) bfr[ks] = load_bfrag(brow + ks * 32 + g * 16);
    v8f acc = {};
    #pragma unroll
    for (int ks = 0; ks < 4; ++ks) acc = wmma16(afr[ks], bfr[ks], acc);
    #pragma unroll
    for (int r = 0; r < 8; ++r) {
      float t = acc[r];
      float ge = 0.5f * t * (1.f + erff(t * 0.70710678118f));   // exact GELU
      hout[(tok0 + r + g * 8) * CDIM + nt * 16 + ln] = (_Float16)ge;
    }
  }
}

// FFN GEMM2 (W2 TDM-staged) + msg residual -> f32 output
__global__ __launch_bounds__(256) void ffn2_kernel(const _Float16* __restrict__ hin,
                                                   const _Float16* __restrict__ w2,
                                                   const float* __restrict__ msg,
                                                   float* __restrict__ out) {
  extern __shared__ _Float16 sW2[];              // 128 x stride 136
  int wave = threadIdx.x >> 5, lane = threadIdx.x & 31, g = lane >> 4, ln = lane & 15;
  size_t tok0 = (size_t)blockIdx.x * 128 + wave * 16;
  if (threadIdx.x < 32) tdm_load_w((unsigned int)(uintptr_t)sW2, w2, 128);
  const _Float16* arow = hin + (tok0 + ln) * CDIM;
  v16h afr[4];
  #pragma unroll
  for (int ks = 0; ks < 4; ++ks) afr[ks] = load_afrag(arow + ks * 32 + g * 8);
  if (threadIdx.x < 32) __builtin_amdgcn_s_wait_tensorcnt(0);
  __syncthreads();
  for (int nt = 0; nt < 8; ++nt) {
    const _Float16* brow = sW2 + (size_t)(nt * 16 + ln) * LDS_STRIDE;
    v16h bfr[4];
    #pragma unroll
    for (int ks = 0; ks < 4; ++ks) bfr[ks] = load_bfrag(brow + ks * 32 + g * 16);
    v8f acc = {};
    #pragma unroll
    for (int ks = 0; ks < 4; ++ks) acc = wmma16(afr[ks], bfr[ks], acc);
    #pragma unroll
    for (int r = 0; r < 8; ++r) {
      size_t t = tok0 + r + g * 8;
      int col = nt * 16 + ln;
      out[t * CDIM + col] = acc[r] + msg[t * CDIM + col];
    }
  }
}

// ---------------- host launcher ----------------
extern "C" void kernel_launch(void* const* d_in, const int* in_sizes, int n_in,
                              void* d_out, int out_size, void* d_ws, size_t ws_size,
                              hipStream_t stream) {
  (void)in_sizes; (void)n_in; (void)out_size; (void)ws_size;
  const float* x  = (const float*)d_in[0];
  const float* Wq = (const float*)d_in[1];
  const float* Wk = (const float*)d_in[2];
  const float* Wv = (const float*)d_in[3];
  const float* Wm = (const float*)d_in[4];
  const float* W1 = (const float*)d_in[5];
  const float* W2 = (const float*)d_in[6];
  const float* g1 = (const float*)d_in[7];
  const float* b1 = (const float*)d_in[8];
  const float* g2 = (const float*)d_in[9];
  const float* b2 = (const float*)d_in[10];
  float* out = (float*)d_out;

  char* ws = (char*)d_ws;
  _Float16* w16   = (_Float16*)(ws + OFF_WQKV);
  _Float16* w1h   = (_Float16*)(ws + OFF_W1);
  _Float16* w2h   = (_Float16*)(ws + OFF_W2);
  _Float16* xn    = (_Float16*)(ws + OFF_XN);
  _Float16* qbuf  = (_Float16*)(ws + OFF_Q);
  float*    part  = (float*)   (ws + OFF_PART);
  float*    kv    = (float*)   (ws + OFF_KV);
  float*    ksum  = (float*)   (ws + OFF_KSUM);
  _Float16* cmat  = (_Float16*)(ws + OFF_CMAT);
  float*    msg   = (float*)   (ws + OFF_MSG);
  _Float16* mn    = xn;      // xn dead after qkv_kernel
  _Float16* hbuf  = qbuf;    // Q dead after msg_kernel

  cvt_w_kernel    <<<320, 256, 0, stream>>>(Wq, Wk, Wv, W1, W2, w16);
  ln_kernel       <<<NTOK / 8, 256, 0, stream>>>(x, g1, b1, xn);
  qkv_kernel      <<<NBLK, 256, (384 + 256) * LDS_STRIDE * 2, stream>>>(xn, w16, qbuf, part);
  reduce_kv_kernel<<<BS * NH, 288, 0, stream>>>(part, kv, ksum);
  make_cmat_kernel<<<BS, 256, 0, stream>>>(kv, Wm, cmat);
  msg_kernel      <<<NBLK, 256, 256 * LDS_STRIDE * 2, stream>>>(qbuf, ksum, cmat, x, g2, b2, msg, mn);
  ffn1_kernel     <<<NBLK, 256, 128 * LDS_STRIDE * 2, stream>>>(mn, w1h, hbuf);
  ffn2_kernel     <<<NBLK, 256, 128 * LDS_STRIDE * 2, stream>>>(hbuf, w2h, msg, out);
}
